// NT_Xent_51780125721381
// MI455X (gfx1250) — matrix operand order repaired and checked by reading, not compiled, not run on hardware
//
#include <hip/hip_runtime.h>
#include <math.h>

typedef _Float16 h16v __attribute__((ext_vector_type(16)));
typedef _Float16 h8v  __attribute__((ext_vector_type(8)));
typedef _Float16 h4v  __attribute__((ext_vector_type(4)));
typedef float    f8v  __attribute__((ext_vector_type(8)));
typedef float    f4v  __attribute__((ext_vector_type(4)));

#define DD 128
#define TEMP_INV 2.0f                         // 1 / TEMP, TEMP = 0.5
#define EXP2_SCALE (TEMP_INV * 1.44269504088896340736f)  // fold /TEMP + log2(e)
#define LN2 0.69314718055994530942f

// ---------------------------------------------------------------------------
// Kernel 1: row-normalize z = concat(z_i, z_j) -> f16 zn.  One wave per row.
// ---------------------------------------------------------------------------
__global__ __launch_bounds__(256) void nt_normalize(
    const float* __restrict__ zi, const float* __restrict__ zj,
    _Float16* __restrict__ zn, int B)
{
    const int wave = threadIdx.x >> 5;
    const int lane = threadIdx.x & 31;
    const int row  = blockIdx.x * (blockDim.x >> 5) + wave;
    const int N    = 2 * B;
    if (row >= N) return;

    const float* src = (row < B) ? (zi + (size_t)row * DD)
                                 : (zj + (size_t)(row - B) * DD);
    f4v v = *(const f4v*)(src + lane * 4);
    float ss = v.x * v.x + v.y * v.y + v.z * v.z + v.w * v.w;
#pragma unroll
    for (int m = 16; m >= 1; m >>= 1) ss += __shfl_xor(ss, m, 32);

    const float inv = 1.0f / fmaxf(sqrtf(ss), 1e-8f);
    h4v o;
    o.x = (_Float16)(v.x * inv);
    o.y = (_Float16)(v.y * inv);
    o.z = (_Float16)(v.z * inv);
    o.w = (_Float16)(v.w * inv);
    *(h4v*)(zn + (size_t)row * DD + lane * 4) = o;
}

// ---------------------------------------------------------------------------
// Kernel 2: positives + diagonal terms.
//   posPart[i]  = (1/TEMP) * dot(zn[i], zn[(i+B)%N])
//   selfPart[i] = exp((1/TEMP) * dot(zn[i], zn[i]))   (the sim_ii exp term,
//                 subtracted from the full row sum in nt_lse instead of
//                 masking inside the hot WMMA loop)
// ---------------------------------------------------------------------------
__global__ __launch_bounds__(256) void nt_pos(
    const _Float16* __restrict__ zn, float* __restrict__ posPart,
    float* __restrict__ selfPart, int B)
{
    const int wave = threadIdx.x >> 5;
    const int lane = threadIdx.x & 31;
    const int i    = blockIdx.x * (blockDim.x >> 5) + wave;
    const int N    = 2 * B;
    if (i >= N) return;
    int p = i + B; if (p >= N) p -= N;

    h4v a = *(const h4v*)(zn + (size_t)i * DD + lane * 4);
    h4v b = *(const h4v*)(zn + (size_t)p * DD + lane * 4);
    float dp = (float)a.x * (float)b.x + (float)a.y * (float)b.y
             + (float)a.z * (float)b.z + (float)a.w * (float)b.w;
    float dd = (float)a.x * (float)a.x + (float)a.y * (float)a.y
             + (float)a.z * (float)a.z + (float)a.w * (float)a.w;
#pragma unroll
    for (int m = 16; m >= 1; m >>= 1) {
        dp += __shfl_xor(dp, m, 32);
        dd += __shfl_xor(dd, m, 32);
    }
    if (lane == 0) {
        posPart[i]  = TEMP_INV * dp;
        selfPart[i] = __builtin_amdgcn_exp2f(dd * EXP2_SCALE);
    }
}

// ---------------------------------------------------------------------------
// Kernel 3: fused WMMA logsumexp.  Each wave owns TWO 16-row tiles (32 rows):
// every B fragment feeds two row tiles (32 FLOP/byte from L2) and the 8 WMMAs
// per iteration form 4 independent 2-deep chains.  The inner loop is pure
// loads -> WMMA -> pk_add/pk_mul -> v_exp -> pk_add accumulate: the diagonal
// is handled by subtracting selfPart[row] once per row before the log.
// ---------------------------------------------------------------------------
__global__ __launch_bounds__(256) void nt_lse(
    const _Float16* __restrict__ zn, const float* __restrict__ selfPart,
    float* __restrict__ lsePart, int B)
{
    const int N       = 2 * B;
    const int nTiles  = N >> 4;
    const int wave    = threadIdx.x >> 5;
    const int lane    = threadIdx.x & 31;
    const int h       = lane >> 4;     // lane half
    const int nl      = lane & 15;
    const int pairIdx = blockIdx.x * 8 + wave;   // which 32-row pair
    const int t0      = pairIdx * 2;             // first 16-row tile
    const int t1      = t0 + 1;                  // second 16-row tile
    const int r0      = t0 * 16;

    // --- A fragments for both row tiles.
    // Row m = nl; half h holds K = 16h..16h+15.
    const _Float16* arow0 = zn + (size_t)(r0 + nl) * DD;
    const _Float16* arow1 = zn + (size_t)(r0 + 16 + nl) * DD;
    h16v afragA[4], afragB[4];
#pragma unroll
    for (int kb = 0; kb < 4; ++kb) {
        h8v lo0 = *(const h8v*)(arow0 + kb * 32 + 16 * h);
        h8v hi0 = *(const h8v*)(arow0 + kb * 32 + 8 + 16 * h);
        afragA[kb] = __builtin_shufflevector(lo0, hi0,
            0, 1, 2, 3, 4, 5, 6, 7, 8, 9, 10, 11, 12, 13, 14, 15);
        h8v lo1 = *(const h8v*)(arow1 + kb * 32 + 16 * h);
        h8v hi1 = *(const h8v*)(arow1 + kb * 32 + 8 + 16 * h);
        afragB[kb] = __builtin_shufflevector(lo1, hi1,
            0, 1, 2, 3, 4, 5, 6, 7, 8, 9, 10, 11, 12, 13, 14, 15);
    }

    f8v accA = {0.f, 0.f, 0.f, 0.f, 0.f, 0.f, 0.f, 0.f};
    f8v accB = {0.f, 0.f, 0.f, 0.f, 0.f, 0.f, 0.f, 0.f};
    const f8v zero = {0.f, 0.f, 0.f, 0.f, 0.f, 0.f, 0.f, 0.f};

    for (int ct = 0; ct < nTiles; ++ct) {
        // B fragment: column n = nl is zn row (ct*16+nl); half h holds K=16h..16h+15
        const _Float16* bcol = zn + (size_t)(ct * 16 + nl) * DD + 16 * h;

        // speculative prefetch of next column tile (OOB prefetch is dropped)
        __builtin_prefetch(bcol + 16 * DD, 0, 3);

        h16v bf0 = *(const h16v*)(bcol);
        h16v bf1 = *(const h16v*)(bcol + 32);
        h16v bf2 = *(const h16v*)(bcol + 64);
        h16v bf3 = *(const h16v*)(bcol + 96);

        // four independent 2-deep WMMA chains
        f8v cA0 = __builtin_amdgcn_wmma_f32_16x16x32_f16(
            false, afragA[0], false, bf0, (short)0, zero, false, false);
        f8v cA1 = __builtin_amdgcn_wmma_f32_16x16x32_f16(
            false, afragA[1], false, bf1, (short)0, zero, false, false);
        f8v cB0 = __builtin_amdgcn_wmma_f32_16x16x32_f16(
            false, afragB[0], false, bf0, (short)0, zero, false, false);
        f8v cB1 = __builtin_amdgcn_wmma_f32_16x16x32_f16(
            false, afragB[1], false, bf1, (short)0, zero, false, false);
        cA0 = __builtin_amdgcn_wmma_f32_16x16x32_f16(
            false, afragA[2], false, bf2, (short)0, cA0, false, false);
        cA1 = __builtin_amdgcn_wmma_f32_16x16x32_f16(
            false, afragA[3], false, bf3, (short)0, cA1, false, false);
        cB0 = __builtin_amdgcn_wmma_f32_16x16x32_f16(
            false, afragB[2], false, bf2, (short)0, cB0, false, false);
        cB1 = __builtin_amdgcn_wmma_f32_16x16x32_f16(
            false, afragB[3], false, bf3, (short)0, cB1, false, false);

        // whole-vector epilogue: keeps v_pk ops on the WMMA register ranges
        f8v sA = (cA0 + cA1) * EXP2_SCALE;   // -> 2^x domain
        f8v sB = (cB0 + cB1) * EXP2_SCALE;
        f8v eA, eB;
#pragma unroll
        for (int v = 0; v < 8; ++v) eA[v] = __builtin_amdgcn_exp2f(sA[v]);
#pragma unroll
        for (int v = 0; v < 8; ++v) eB[v] = __builtin_amdgcn_exp2f(sB[v]);

        accA += eA;
        accB += eB;
    }

    // reduce the 16 lanes of each half (each lane holds disjoint column groups)
    float sumsA[8], sumsB[8];
#pragma unroll
    for (int v = 0; v < 8; ++v) { sumsA[v] = accA[v]; sumsB[v] = accB[v]; }
#pragma unroll
    for (int v = 0; v < 8; ++v) {
#pragma unroll
        for (int m = 1; m < 16; m <<= 1) {
            sumsA[v] += __shfl_xor(sumsA[v], m, 32);
            sumsB[v] += __shfl_xor(sumsB[v], m, 32);
        }
    }

    if (nl == 0) {  // lanes 0 and 16: rows base+8h+{0..7} of each tile
        float lA = 0.0f, lB = 0.0f;
#pragma unroll
        for (int v = 0; v < 8; ++v) {
            // exclude the diagonal term, then take the log (v_log_f32 = log2)
            float sA = sumsA[v] - selfPart[r0 + 8 * h + v];
            float sB = sumsB[v] - selfPart[r0 + 16 + 8 * h + v];
            lA += __builtin_amdgcn_logf(sA);
            lB += __builtin_amdgcn_logf(sB);
        }
        lsePart[t0 * 2 + h] = lA * LN2;
        lsePart[t1 * 2 + h] = lB * LN2;
    }
}

// ---------------------------------------------------------------------------
// Kernel 4: deterministic single-block reduction -> loss scalar.
// ---------------------------------------------------------------------------
__global__ __launch_bounds__(256) void nt_finalize(
    const float* __restrict__ lsePart, const float* __restrict__ posPart,
    float* __restrict__ out, int nLse, int nPos, int N)
{
    __shared__ float sm[256];
    float t = 0.0f;
    for (int i = threadIdx.x; i < nLse; i += 256) t += lsePart[i];
    for (int i = threadIdx.x; i < nPos; i += 256) t -= posPart[i];
    sm[threadIdx.x] = t;
    __syncthreads();
    for (int s = 128; s > 0; s >>= 1) {
        if ((int)threadIdx.x < s) sm[threadIdx.x] += sm[threadIdx.x + s];
        __syncthreads();
    }
    if (threadIdx.x == 0) out[0] = sm[0] / (float)N;
}

// ---------------------------------------------------------------------------
extern "C" void kernel_launch(void* const* d_in, const int* in_sizes, int n_in,
                              void* d_out, int out_size, void* d_ws, size_t ws_size,
                              hipStream_t stream)
{
    (void)n_in; (void)out_size; (void)ws_size;
    const float* zi = (const float*)d_in[0];
    const float* zj = (const float*)d_in[1];
    const int B = in_sizes[0] / DD;   // 4096
    const int N = 2 * B;              // 8192

    // workspace layout:
    //   [0, 4KB)        lsePart  : N/16 * 2 floats  (1024 floats)
    //   [4KB, 36KB)     posPart  : N floats         (8192 floats)
    //   [36KB, 68KB)    selfPart : N floats         (8192 floats)
    //   [128KB, ...)    zn       : N*DD f16         (2 MB)
    char* ws = (char*)d_ws;
    float*    lsePart  = (float*)(ws);
    float*    posPart  = (float*)(ws + 4096);
    float*    selfPart = (float*)(ws + 36864);
    _Float16* zn       = (_Float16*)(ws + 131072);
    float*    out      = (float*)d_out;

    const int wavesPerBlk = 8;                   // 256 threads, wave32
    nt_normalize<<<N / wavesPerBlk, 256, 0, stream>>>(zi, zj, zn, B);
    nt_pos      <<<N / wavesPerBlk, 256, 0, stream>>>(zn, posPart, selfPart, B);
    nt_lse      <<<N / (32 * wavesPerBlk), 256, 0, stream>>>(zn, selfPart, lsePart, B);
    nt_finalize <<<1, 256, 0, stream>>>(lsePart, posPart, out,
                                        (N / 16) * 2, N, N);
}
